// Attention_46643344834850
// MI455X (gfx1250) — matrix operand order repaired and checked by reading, not compiled, not run on hardware
//
#include <hip/hip_runtime.h>
#include <hip/hip_bf16.h>
#include <math.h>

typedef _Float16 h16;
typedef __attribute__((ext_vector_type(16))) _Float16 v16h;
typedef __attribute__((ext_vector_type(8)))  float    v8f;
typedef __attribute__((ext_vector_type(4)))  unsigned int u32x4;
typedef __attribute__((ext_vector_type(8)))  int          i32x8;
typedef __attribute__((ext_vector_type(4)))  int          i32x4;

union AF { v16h v; h16 h[16]; };
union CF { v8f  v; float f[8]; };

#define B_    8
#define C_    512
#define NH_   1024
#define AH_   32
#define HEADS 8
#define DH    64

// ---- fragment index helpers (contiguous k order of this file, wave32) ----
// A-matrix 16x32 f16: lane holds row M=lane&15; half-slot j -> K:
//   K = j ; +16 for lanes 16-31 (contiguous, matches B fills)
__device__ __forceinline__ int kA(int j, int lane) {
  return j + ((lane & 16) ? 16 : 0);
}

__device__ __forceinline__ v8f wmma_f16(v16h a, v16h b, v8f c) {
  return __builtin_amdgcn_wmma_f32_16x16x32_f16(false, a, false, b, (short)0, c,
                                                false, false);
}

// =====================  DFT passes (global filter)  =====================
// thread id decomposition: c[0:8] j[9:13] i[14:18] b[19:21]  (4,194,304 threads)

// forward DFT along width (w -> v).  branch 0: q [B,1024,C]; branch 1: x [B,256,C] NN-upsampled
__global__ void gf_row_fwd(const float* __restrict__ src, float* __restrict__ dst,
                           int branch) {
  int tid = blockIdx.x * blockDim.x + threadIdx.x;
  int c = tid & (C_ - 1);
  int v = (tid >> 9) & 31;
  int i = (tid >> 14) & 31;
  int b = tid >> 19;
  float re = 0.f, im = 0.f;
  for (int w = 0; w < 32; ++w) {
    float t;
    if (branch == 0) t = src[(b * NH_ + i * 32 + w) * C_ + c];
    else             t = src[(b * 256 + (i >> 1) * 16 + (w >> 1)) * C_ + c];
    float s, co;
    __sincosf(0.19634954f * (float)((v * w) & 31), &s, &co);
    re += t * co;
    im -= t * s;
  }
  int o = (((b * AH_ + i) * AH_ + v) * C_ + c) * 2;
  dst[o] = re; dst[o + 1] = im;
}

// forward DFT along height (h -> u) + Hermitian-extended weight mul + 1/1024 norm
__global__ void gf_col_fwd_mul(const float* __restrict__ src,
                               const float* __restrict__ cw,
                               float* __restrict__ dst) {
  int tid = blockIdx.x * blockDim.x + threadIdx.x;
  int c = tid & (C_ - 1);
  int v = (tid >> 9) & 31;
  int u = (tid >> 14) & 31;
  int b = tid >> 19;
  float re = 0.f, im = 0.f;
  for (int hh = 0; hh < 32; ++hh) {
    int o = (((b * AH_ + hh) * AH_ + v) * C_ + c) * 2;
    float ar = src[o], ai = src[o + 1];
    float s, co;
    __sincosf(0.19634954f * (float)((u * hh) & 31), &s, &co);
    re += ar * co + ai * s;   // * e^{-i theta}
    im += ai * co - ar * s;
  }
  float wr, wi;
  if (v <= 16) {
    int o = ((u * 17 + v) * C_ + c) * 2;
    wr = cw[o]; wi = cw[o + 1];
  } else {                       // W_full[u,v] = conj(W[(32-u)%32, 32-v])
    int uu = (32 - u) & 31, vv = 32 - v;
    int o = ((uu * 17 + vv) * C_ + c) * 2;
    wr = cw[o]; wi = -cw[o + 1];
  }
  const float n = 1.0f / 1024.0f;   // ortho fwd * ortho inv
  int o2 = (((b * AH_ + u) * AH_ + v) * C_ + c) * 2;
  dst[o2]     = (re * wr - im * wi) * n;
  dst[o2 + 1] = (re * wi + im * wr) * n;
}

// inverse DFT along height (u -> h)
__global__ void gf_col_inv(const float* __restrict__ src, float* __restrict__ dst) {
  int tid = blockIdx.x * blockDim.x + threadIdx.x;
  int c = tid & (C_ - 1);
  int v = (tid >> 9) & 31;
  int i = (tid >> 14) & 31;
  int b = tid >> 19;
  float re = 0.f, im = 0.f;
  for (int u = 0; u < 32; ++u) {
    int o = (((b * AH_ + u) * AH_ + v) * C_ + c) * 2;
    float ar = src[o], ai = src[o + 1];
    float s, co;
    __sincosf(0.19634954f * (float)((u * i) & 31), &s, &co);
    re += ar * co - ai * s;   // * e^{+i theta}
    im += ai * co + ar * s;
  }
  int o2 = (((b * AH_ + i) * AH_ + v) * C_ + c) * 2;
  dst[o2] = re; dst[o2 + 1] = im;
}

// inverse DFT along width (v -> w), take real, pack per-head f16 [B,heads,1024,dh]
__global__ void gf_row_inv_pack(const float* __restrict__ src, h16* __restrict__ dst) {
  int tid = blockIdx.x * blockDim.x + threadIdx.x;
  int c = tid & (C_ - 1);
  int w = (tid >> 9) & 31;
  int i = (tid >> 14) & 31;
  int b = tid >> 19;
  float re = 0.f;
  for (int v = 0; v < 32; ++v) {
    int o = (((b * AH_ + i) * AH_ + v) * C_ + c) * 2;
    float ar = src[o], ai = src[o + 1];
    float s, co;
    __sincosf(0.19634954f * (float)((v * w) & 31), &s, &co);
    re += ar * co - ai * s;
  }
  int head = c & (HEADS - 1);
  int d = c >> 3;                 // c = d*heads + head
  dst[((b * HEADS + head) * NH_ + i * 32 + w) * DH + d] = (h16)re;
}

// ==================  fused sigmoid attention (WMMA + TDM)  ==============
// block = 256 threads = 8 waves; block handles (b, head, 128 q-rows)
__global__ void __launch_bounds__(256)
attn_sigmoid_wmma(const h16* __restrict__ qh, const h16* __restrict__ kvh,
                  h16* __restrict__ ctx) {
  __shared__ __align__(16) h16 Kt[32 * 64];      // K/V tile, row-major (TDM dest)
  __shared__ __align__(16) h16 VtT[64 * 32];     // transposed copy for PV B-frags
  __shared__ float sP[8][16 * 32];               // per-wave prob tile re-stripe

  const int lane = threadIdx.x & 31;
  const int wv   = threadIdx.x >> 5;
  const int blk  = blockIdx.x;
  const int grp  = blk & 7;
  const int hd   = (blk >> 3) & 7;
  const int b    = blk >> 6;

  const h16* Qb = qh  + (size_t)(b * HEADS + hd) * NH_ * DH;
  const h16* Kb = kvh + (size_t)(b * HEADS + hd) * NH_ * DH;
  const int i0 = grp * 128 + wv * 16;

  // load this wave's Q rows as two A fragments (K-dim chunks 0-31 / 32-63)
  AF qa0, qa1;
  {
    const h16* qrow = Qb + (i0 + (lane & 15)) * DH;
    for (int j = 0; j < 16; ++j) {
      int k = kA(j, lane);
      qa0.h[j] = qrow[k];
      qa1.h[j] = qrow[32 + k];
    }
  }

  CF acc[4];
  for (int nc = 0; nc < 4; ++nc)
    for (int v = 0; v < 8; ++v) acc[nc].f[v] = 0.f;

  for (int j0 = 0; j0 < NH_; j0 += 32) {
    __syncthreads();
    // ---- TDM: DMA the 32x64 f16 K/V tile into LDS (one wave issues) ----
    if (wv == 0) {
      unsigned long long ga = (unsigned long long)(uintptr_t)(Kb + (size_t)j0 * DH);
      unsigned int lds_off = (unsigned int)(uintptr_t)(&Kt[0]);  // LDS byte offset
      u32x4 g0;
      g0.x = 1u;                                        // count=1 valid, no gather
      g0.y = lds_off;                                   // lds_addr
      g0.z = (unsigned int)ga;                          // global_addr[31:0]
      g0.w = (unsigned int)((ga >> 32) & 0x1FFFFFFu) | (2u << 30);  // [56:32], type=2
      i32x8 g1;
      g1[0] = (1 << 16);          // workgroup_mask=0, data_size=1 (2 bytes)
      g1[1] = (DH << 16);         // tensor_dim0 = 64 elems (bits 63:48)
      g1[2] = (NH_ << 16);        // tensor_dim1 = 1024 (bits 95:80)
      g1[3] = (DH << 16);         // tile_dim0 = 64 (bits 127:112)
      g1[4] = 32;                 // tile_dim1 = 32, tile_dim2 = 0
      g1[5] = DH;                 // tensor_dim0_stride = 64
      g1[6] = 0;
      g1[7] = 0;
      i32x4 z4 = {0, 0, 0, 0};            // groups 2/3 unused (2-D tensor)
      i32x8 z8 = {0, 0, 0, 0, 0, 0, 0, 0};
      __builtin_amdgcn_tensor_load_to_lds(g0, g1, z4, z4, z8, 0);
      __builtin_amdgcn_s_wait_tensorcnt(0);
    }
    // prefetch next K tile toward L2 while this one is consumed
    __builtin_prefetch(Kb + (size_t)((j0 + 32) & (NH_ - 1)) * DH, 0, 1);
    __syncthreads();

    // ---- cooperative transpose Kt -> VtT (contiguous PV fragment reads) ----
    {
      int t = threadIdx.x;
      int row = t >> 3, col = (t & 7) * 8;
      h16 tmp[8];
      *(uint4*)tmp = *(const uint4*)(&Kt[row * 64 + col]);
      for (int i = 0; i < 8; ++i) VtT[(col + i) * 32 + row] = tmp[i];
    }
    __syncthreads();

    // ---- scores: two 16x16 tiles, K-dim 64 = 2 WMMA each ----
    CF s1, s2;
    for (int v = 0; v < 8; ++v) { s1.f[v] = 0.f; s2.f[v] = 0.f; }
    {
      int n = lane & 15;
      int kofs = (lane & 16) ? 16 : 0;
      AF kb;
      for (int j = 0; j < 16; ++j) kb.h[j] = Kt[n * 64 + kofs + j];
      s1.v = wmma_f16(qa0.v, kb.v, s1.v);
      for (int j = 0; j < 16; ++j) kb.h[j] = Kt[n * 64 + 32 + kofs + j];
      s1.v = wmma_f16(qa1.v, kb.v, s1.v);
      for (int j = 0; j < 16; ++j) kb.h[j] = Kt[(16 + n) * 64 + kofs + j];
      s2.v = wmma_f16(qa0.v, kb.v, s2.v);
      for (int j = 0; j < 16; ++j) kb.h[j] = Kt[(16 + n) * 64 + 32 + kofs + j];
      s2.v = wmma_f16(qa1.v, kb.v, s2.v);
    }

    // ---- sigmoid(s/8) -> per-wave LDS (C-layout -> A-layout re-stripe) ----
    {
      int col = lane & 15;
      int rofs = (lane & 16) ? 8 : 0;
      for (int v = 0; v < 8; ++v) {
        sP[wv][(v + rofs) * 32 + col]      = 1.f / (1.f + __expf(-0.125f * s1.f[v]));
        sP[wv][(v + rofs) * 32 + 16 + col] = 1.f / (1.f + __expf(-0.125f * s2.f[v]));
      }
    }
    AF pa;
    {
      int m = lane & 15;
      for (int j = 0; j < 16; ++j) pa.h[j] = (h16)sP[wv][m * 32 + kA(j, lane)];
    }

    // ---- ctx += P(16x32) * V(32x64), 4 column chunks (contiguous VtT rows) ----
    {
      int n = lane & 15;
      int kofs = (lane & 16) ? 16 : 0;
      for (int nc = 0; nc < 4; ++nc) {
        AF vb;
        const h16* vt = &VtT[(nc * 16 + n) * 32 + kofs];
        for (int j = 0; j < 16; ++j) vb.h[j] = vt[j];
        acc[nc].v = wmma_f16(pa.v, vb.v, acc[nc].v);
      }
    }
  }

  // write ctx f16 back in channel-interleaved layout c = d*heads + head
  {
    int n = lane & 15;
    int rofs = (lane & 16) ? 8 : 0;
    for (int nc = 0; nc < 4; ++nc)
      for (int v = 0; v < 8; ++v) {
        int m = i0 + v + rofs;
        int d = nc * 16 + n;
        ctx[(size_t)(b * NH_ + m) * C_ + d * HEADS + hd] = (h16)acc[nc].f[v];
      }
  }
}

// ==================  output projection (WMMA GEMM + bias)  ==============
// out[8192,512] = ctx[8192,512](f16) * out_w^T(f32->f16) + bias; wave = 16x64 tile
__global__ void __launch_bounds__(256)
proj_wmma(const h16* __restrict__ ctx, const float* __restrict__ w,
          const float* __restrict__ bias, float* __restrict__ out) {
  const int lane = threadIdx.x & 31;
  const int wv   = threadIdx.x >> 5;
  const int gw   = blockIdx.x * 8 + wv;      // 4096 wave-tiles
  const int nt   = gw & 7;
  const int mt   = gw >> 3;
  const int m0 = mt * 16, n0 = nt * 64;

  CF acc[4];
  for (int nc = 0; nc < 4; ++nc)
    for (int v = 0; v < 8; ++v) acc[nc].f[v] = 0.f;

  const int mn = lane & 15;                  // row for A, col for B
  const int kofs = (lane & 16) ? 16 : 0;
  for (int k0 = 0; k0 < C_; k0 += 32) {
    AF a;
    const h16* crow = ctx + (size_t)(m0 + mn) * C_ + k0;
    for (int j = 0; j < 16; ++j) a.h[j] = crow[kA(j, lane)];
    for (int nc = 0; nc < 4; ++nc) {
      AF bf;
      const float* wrow = w + (size_t)(n0 + nc * 16 + mn) * C_ + k0 + kofs;
      for (int j = 0; j < 16; ++j) bf.h[j] = (h16)wrow[j];
      acc[nc].v = wmma_f16(a.v, bf.v, acc[nc].v);
    }
  }

  const int rofs = (lane & 16) ? 8 : 0;
  for (int nc = 0; nc < 4; ++nc) {
    int col = n0 + nc * 16 + (lane & 15);
    float bb = bias[col];
    for (int v = 0; v < 8; ++v)
      out[(size_t)(m0 + v + rofs) * C_ + col] = acc[nc].f[v] + bb;
  }
}

// ============================  launcher  ================================
extern "C" void kernel_launch(void* const* d_in, const int* in_sizes, int n_in,
                              void* d_out, int out_size, void* d_ws, size_t ws_size,
                              hipStream_t stream) {
  const float* q  = (const float*)d_in[0];   // [8,1024,512]
  const float* x  = (const float*)d_in[1];   // [8,256,512]
  const float* cw = (const float*)d_in[2];   // [32,17,512,2]
  const float* ow = (const float*)d_in[3];   // [512,512]
  const float* ob = (const float*)d_in[4];   // [512]
  float* out = (float*)d_out;                // [8,1024,512] f32

  char* ws = (char*)d_ws;
  float* cb1 = (float*)(ws);                          // 33,554,432 B complex
  float* cb2 = (float*)(ws + 33554432);               // 33,554,432 B complex
  h16*   qh  = (h16*)(ws + 67108864);                 // 8,388,608 B f16
  h16*   xh  = (h16*)(ws + 75497472);                 // 8,388,608 B f16
  h16*   ctx = (h16*)(ws + 83886080);                 // 8,388,608 B f16

  dim3 blk(256);
  const int dftBlocks = (B_ * AH_ * AH_ * C_) / 256;  // 16384

  // q branch global filter -> per-head f16 Q
  gf_row_fwd     <<<dftBlocks, blk, 0, stream>>>(q, cb1, 0);
  gf_col_fwd_mul <<<dftBlocks, blk, 0, stream>>>(cb1, cw, cb2);
  gf_col_inv     <<<dftBlocks, blk, 0, stream>>>(cb2, cb1);
  gf_row_inv_pack<<<dftBlocks, blk, 0, stream>>>(cb1, qh);

  // x branch (NN upsample fused into first pass) -> per-head f16 K/V
  gf_row_fwd     <<<dftBlocks, blk, 0, stream>>>(x, cb1, 1);
  gf_col_fwd_mul <<<dftBlocks, blk, 0, stream>>>(cb1, cw, cb2);
  gf_col_inv     <<<dftBlocks, blk, 0, stream>>>(cb2, cb1);
  gf_row_inv_pack<<<dftBlocks, blk, 0, stream>>>(cb1, xh);

  // fused sigmoid attention: 512 blocks x (b,head,128 rows), TDM-staged K/V
  attn_sigmoid_wmma<<<512, blk, 0, stream>>>(qh, xh, ctx);

  // output projection + bias
  proj_wmma<<<512, blk, 0, stream>>>(ctx, ow, ob, out);
}
